// LiquidBlock_47476568490054
// MI455X (gfx1250) — matrix operand order repaired and checked by reading, not compiled, not run on hardware
//
#include <hip/hip_runtime.h>

// ---------------------------------------------------------------------------
// MI455X (gfx1250, wave32) LiquidBlock forward.
// Matrix ops: V_WMMA_F32_16X16X32_BF16 (bf16 in, f32 accumulate).
// GEMM blocks: M=64 x N=64 per 4-wave block; 4 accumulators/wave so each
// B-fragment feeds 4 WMMAs; A panels staged in double-buffered 64x96 LDS
// chunks (one barrier per 12 WMMA/wave). A-fragments batch-loaded per k-step
// so one s_wait_dscnt covers 4 WMMAs.
// ---------------------------------------------------------------------------

typedef __bf16 bf16;
typedef bf16  v16bf __attribute__((ext_vector_type(16)));
typedef bf16  v8bf  __attribute__((ext_vector_type(8)));
typedef float v8f   __attribute__((ext_vector_type(8)));

#define EPI_BF16   0   // outB = act(acc + bias)
#define EPI_F32ADD 1   // outF = acc + bias + src
#define EPI_DUAL   2   // base=src; outB=bf16(base+acc+bias); outF=base+(acc+bias)*scale

#define KC 96          // K-chunk (divides 192, 288, 384, 768, 1152)

static __device__ __forceinline__ v8f wmma_bf(v16bf a, v16bf b, v8f c) {
  return __builtin_amdgcn_wmma_f32_16x16x32_bf16(false, a, false, b, (short)0, c,
                                                 false, false);
}

// A-fragment from an LDS panel (row-major, stride ld, 16 rows) at column koff.
// k order used here (contiguous, as in load_b_frag): lane<16 -> row=lane, K=koff..koff+15;
//             lane>=16 -> row=lane-16, K=koff+16..koff+31.
static __device__ __forceinline__ v16bf load_a_frag(const bf16* As, int ld, int koff) {
  int lane = threadIdx.x & 31;
  const bf16* p = As + (lane & 15) * ld + koff + ((lane >> 4) << 4);
  v8bf lo = *(const v8bf*)p;
  v8bf hi = *(const v8bf*)(p + 8);
  return __builtin_shufflevector(lo, hi, 0,1,2,3,4,5,6,7,8,9,10,11,12,13,14,15);
}

// B-fragment from N-major transposed weight Wt (N x K, row-major bf16).
// Contiguous k (as in load_a_frag): lane<16 -> col=n0+lane, K=k0..k0+15; lane>=16 -> K=k0+16..k0+31.
static __device__ __forceinline__ v16bf load_b_frag(const bf16* Wt, int ldk,
                                                    int n0, int k0) {
  int lane = threadIdx.x & 31;
  const bf16* p = Wt + (long long)(n0 + (lane & 15)) * ldk + k0 + ((lane >> 4) << 4);
  return *(const v16bf*)p;
}

// Stage a 64 x 96 bf16 panel: 128 threads x six 16B pieces.
static __device__ __forceinline__ void stage_a64(bf16* dst, const bf16* src,
                                                 int lda) {
#pragma unroll
  for (int it = 0; it < 6; ++it) {
    int e = (threadIdx.x + it * 128) << 3;   // 8 bf16 per piece
    int r = e / KC, c = e % KC;
    *(v8bf*)(dst + r * KC + c) = *(const v8bf*)(src + (long long)r * lda + c);
  }
}

static __device__ __forceinline__ float sigm(float v) {
  return 1.f / (1.f + __expf(-v));
}

// ---------------------------------------------------------------------------
// Weight prep: W (K x N, f32, row stride ldw) -> Wt (N x K, bf16)
// ---------------------------------------------------------------------------
__global__ void k_wt(const float* __restrict__ W, int ldw,
                     bf16* __restrict__ Wt, int K, int N) {
  long long idx = (long long)blockIdx.x * blockDim.x + threadIdx.x;
  if (idx >= (long long)N * K) return;
  int n = (int)(idx / K), k = (int)(idx % K);
  Wt[idx] = (bf16)W[(long long)k * ldw + n];
}

// Fused GRU weight: Wtg[band][n<576][k<384]; k<192 -> Wx[k][n];
// k>=192 -> (n<384 ? Wh[k-192][n] : 0)   (hc column block excluded)
__global__ void k_wt_gru(const float* __restrict__ Wx,
                         const float* __restrict__ Wh,
                         bf16* __restrict__ Wtg) {
  long long idx = (long long)blockIdx.x * blockDim.x + threadIdx.x;
  if (idx >= 4LL * 576 * 384) return;
  int band = (int)(idx / (576 * 384));
  int r = (int)(idx % (576 * 384));
  int n = r / 384, k = r % 384;
  float v;
  if (k < 192)      v = Wx[(long long)band * 110592 + (long long)k * 576 + n];
  else if (n < 384) v = Wh[(long long)band * 110592 + (long long)(k - 192) * 576 + n];
  else              v = 0.f;
  Wtg[idx] = (bf16)v;
}

// gbias[band][0:192]=bx_z+bh_z, [192:384]=bx_r+bh_r, [384:576]=bx_c, [576:768]=bh_c
__global__ void k_gbias(const float* __restrict__ bx, const float* __restrict__ bh,
                        float* __restrict__ gb) {
  int idx = blockIdx.x * blockDim.x + threadIdx.x;
  if (idx >= 4 * 768) return;
  int band = idx / 768, j = idx % 768;
  float v;
  if (j < 384)      v = bx[band * 576 + j] + bh[band * 576 + j];
  else if (j < 576) v = bx[band * 576 + j];
  else              v = bh[band * 576 + (j - 192)];
  gb[idx] = v;
}

// ---------------------------------------------------------------------------
// Row LayerNorm: X (rows x cols, f32) -> Y bf16.  gmode=1: per-band gamma/beta
// indexed by (row % 4). One wave per row.
// ---------------------------------------------------------------------------
__global__ void k_ln(const float* __restrict__ X, bf16* __restrict__ Y,
                     int cols, int gmode,
                     const float* __restrict__ g, const float* __restrict__ b,
                     int rows) {
  int row = blockIdx.x * (blockDim.x >> 5) + (threadIdx.x >> 5);
  if (row >= rows) return;
  int lane = threadIdx.x & 31;
  const float* x = X + (long long)row * cols;
  float s = 0.f, s2 = 0.f;
  for (int c = lane; c < cols; c += 32) { float v = x[c]; s += v; s2 += v * v; }
  for (int o = 16; o > 0; o >>= 1) { s += __shfl_xor(s, o, 32); s2 += __shfl_xor(s2, o, 32); }
  float mean = s / cols;
  float var  = s2 / cols - mean * mean;
  float inv  = rsqrtf(var + 1e-5f);
  int gofs = gmode ? (row & 3) * cols : 0;
  bf16* y = Y + (long long)row * cols;
  for (int c = lane; c < cols; c += 32)
    y[c] = (bf16)((x[c] - mean) * inv * g[gofs + c] + b[gofs + c]);
}

// Pack A_gru[b][band*384 + j] = j<192 ? bi[b][band*192+j] : bf16(h[b][band*192+j-192])
__global__ void k_pack(const bf16* __restrict__ bi, const float* __restrict__ h,
                       bf16* __restrict__ Ag, int B) {
  long long idx = (long long)blockIdx.x * blockDim.x + threadIdx.x;
  if (idx >= (long long)B * 1536) return;
  long long b = idx / 1536;
  int j = (int)(idx % 1536);
  int band = j / 384, jj = j % 384;
  Ag[idx] = (jj < 192) ? bi[b * 768 + band * 192 + jj]
                       : (bf16)h[b * 768 + band * 192 + (jj - 192)];
}

// ---------------------------------------------------------------------------
// Generic WMMA GEMM: C(M x N) = A(M x K, bf16) @ Wt^T + bias.
// Block = 4 waves; tile M=64 x N=64; wave owns one N-16 column and 4 M-tiles.
// A staged in double-buffered 64xKC LDS chunks. grid.z = band.
// ---------------------------------------------------------------------------
__global__ __launch_bounds__(128) void k_gemm(
    const bf16* __restrict__ A, int lda, int aBandOff,
    const bf16* __restrict__ Wt, int wtBandOff,
    const float* __restrict__ bias, int biasBandOff,
    int M, int N, int K, int mode, int act,
    float* __restrict__ outF, int ldoF, int oFBandOff,
    bf16* __restrict__ outB, int ldoB, int oBBandOff,
    const float* __restrict__ src, int ldsrc, int srcBandOff,
    const float* __restrict__ scalep) {
  __shared__ __align__(16) bf16 As[2][64 * KC];
  const int z = blockIdx.z;
  const bf16* Az = A + (long long)z * aBandOff + (long long)blockIdx.x * 64 * lda;
  const bf16* Wz = Wt + (long long)z * wtBandOff;
  const int m0 = blockIdx.x * 64;
  const int wave = threadIdx.x >> 5;
  const int lane = threadIdx.x & 31;
  int n0 = blockIdx.y * 64 + wave * 16;
  if (n0 + 16 > N) n0 = N - 16;           // clamp: duplicate tile, benign
  v8f acc[4] = {};
  const int nch = K / KC;
  stage_a64(As[0], Az, lda);
  for (int c = 0; c < nch; ++c) {
    __syncthreads();
    if (c + 1 < nch) stage_a64(As[(c + 1) & 1], Az + (c + 1) * KC, lda);
    const bf16* ap = As[c & 1];
    const int kc = c * KC;
#pragma unroll
    for (int kk = 0; kk < KC / 32; ++kk) {
      v16bf b = load_b_frag(Wz, K, n0, kc + kk * 32);
      v16bf a[4];
#pragma unroll
      for (int mt = 0; mt < 4; ++mt)       // batch the 8 ds_load_b128
        a[mt] = load_a_frag(ap + mt * 16 * KC, KC, kk * 32);
#pragma unroll
      for (int mt = 0; mt < 4; ++mt)       // then 4 back-to-back WMMAs
        acc[mt] = wmma_bf(a[mt], b, acc[mt]);
    }
  }
  // Epilogue. C layout: vgpr i -> row = mtile + (lane>>4)*8 + i, col = n0+(lane&15)
  int col = n0 + (lane & 15);
  float bval = bias ? bias[(long long)z * biasBandOff + col] : 0.f;
  float scale = scalep ? *scalep : 1.f;
#pragma unroll
  for (int mt = 0; mt < 4; ++mt) {
    int rbase = m0 + mt * 16 + ((lane >> 4) << 3);
#pragma unroll
    for (int i = 0; i < 8; ++i) {
      int row = rbase + i;
      float v = acc[mt][i] + bval;
      if (mode == EPI_BF16) {
        if (act == 1) v = v * sigm(v);     // SiLU
        outB[(long long)row * ldoB + (long long)z * oBBandOff + col] = (bf16)v;
      } else if (mode == EPI_F32ADD) {
        float s = src ? src[(long long)row * ldsrc + (long long)z * srcBandOff + col] : 0.f;
        outF[(long long)row * ldoF + (long long)z * oFBandOff + col] = v + s;
      } else {                             // EPI_DUAL (band FFN computed once)
        float b0 = src[(long long)row * ldsrc + (long long)z * srcBandOff + col];
        outB[(long long)row * ldoB + (long long)z * oBBandOff + col] = (bf16)(b0 + v);
        outF[(long long)row * ldoF + (long long)z * oFBandOff + col] = b0 + v * scale;
      }
    }
  }
}

// ---------------------------------------------------------------------------
// Fused GRU: per (band, 16-row tile) compute P(16x576)=[bi|h]@[[Wx],[Wh(z,r),0]]
// and Q(16x192)=h@Wh_c. Full 16x384 A panel staged once (single barrier),
// then 12 barrier-free K-steps; 48 WMMA tiles over 8 waves; gates in epilogue.
// ---------------------------------------------------------------------------
__global__ __launch_bounds__(256) void k_gru(
    const bf16* __restrict__ Ag,          // B x 1536 ([bi|h] per band)
    const bf16* __restrict__ Wtg,         // 4 x 576 x 384
    const bf16* __restrict__ Wthc,        // 4 x 192 x 192
    const float* __restrict__ gbias,      // 4 x 768
    const float* __restrict__ hstate,     // B x 768 (f32)
    float* __restrict__ hnew) {           // B x 768 (f32)
  __shared__ __align__(16) bf16 As[16 * 384];
  __shared__ float Ps[16 * 768];
  const int band = blockIdx.z;
  const int m0 = blockIdx.x * 16;
  const int wave = threadIdx.x >> 5;
  const int lane = threadIdx.x & 31;
  const bf16* Wg_ = Wtg + band * (576 * 384);
  const bf16* Wh_ = Wthc + band * (192 * 192);
  const bf16* Asrc = Ag + (long long)m0 * 1536 + band * 384;
#pragma unroll
  for (int it = 0; it < 3; ++it) {         // stage 16x384 panel: 256 thr x 3x16B
    int e = (threadIdx.x + it * 256) << 3;
    int r = e / 384, c = e % 384;
    *(v8bf*)(As + r * 384 + c) = *(const v8bf*)(Asrc + (long long)r * 1536 + c);
  }
  __syncthreads();
  v8f acc[6] = {};
  if (wave < 6) {                          // P tiles: cols 0..575, K=384
#pragma unroll
    for (int k0 = 0; k0 < 384; k0 += 32) {
      v16bf a = load_a_frag(As, 384, k0);
      v16bf b[6];
#pragma unroll
      for (int i = 0; i < 6; ++i)
        b[i] = load_b_frag(Wg_, 384, (wave * 6 + i) * 16, k0);
#pragma unroll
      for (int i = 0; i < 6; ++i) acc[i] = wmma_bf(a, b[i], acc[i]);
    }
  } else {                                 // Q tiles: hc = h @ Wh_c, K=192
#pragma unroll
    for (int k0 = 0; k0 < 192; k0 += 32) {
      v16bf a = load_a_frag(As, 384, 192 + k0);  // h half of [bi|h]
      v16bf b[6];
#pragma unroll
      for (int i = 0; i < 6; ++i)
        b[i] = load_b_frag(Wh_, 192, ((wave - 6) * 6 + i) * 16, k0);
#pragma unroll
      for (int i = 0; i < 6; ++i) acc[i] = wmma_bf(a, b[i], acc[i]);
    }
  }
  __syncthreads();
#pragma unroll
  for (int i = 0; i < 6; ++i) {            // spill accumulators: Ps = [P | Q]
    int ct = (wave < 6) ? (wave * 6 + i) : (36 + (wave - 6) * 6 + i);
    int colp = ct * 16 + (lane & 15);
    int rp = (lane >> 4) << 3;
#pragma unroll
    for (int j = 0; j < 8; ++j) Ps[(rp + j) * 768 + colp] = acc[i][j];
  }
  __syncthreads();
  const float* gb = gbias + band * 768;
  for (int e = threadIdx.x; e < 16 * 192; e += 256) {
    int mr = e / 192, s = e % 192;
    float zz = Ps[mr * 768 + s] + gb[s];
    float rr = Ps[mr * 768 + 192 + s] + gb[192 + s];
    float xc = Ps[mr * 768 + 384 + s] + gb[384 + s];
    float hc = Ps[mr * 768 + 576 + s] + gb[576 + s];
    float h = hstate[(long long)(m0 + mr) * 768 + band * 192 + s];
    float zg = sigm(zz), rg = sigm(rr);
    float nn = tanhf(xc + rg * hc);
    hnew[(long long)(m0 + mr) * 768 + band * 192 + s] = (1.f - zg) * nn + zg * h;
  }
}

// ---------------------------------------------------------------------------
// Cross-band attention: NB=4 positions, H=4 heads, DH=48. One thread per
// (b, band, head). ~6 kFLOP/row -> VALU, not worth WMMA.
// ---------------------------------------------------------------------------
__global__ void k_attn(const bf16* __restrict__ qkv,  // (4B) x 576: [q|k|v]
                       bf16* __restrict__ ao, int B) {
  long long t = (long long)blockIdx.x * blockDim.x + threadIdx.x;
  if (t >= (long long)B * 16) return;
  int h = (int)(t & 3), n = (int)((t >> 2) & 3);
  long long b = t >> 4;
  const bf16* base = qkv + b * 4 * 576;
  const bf16* q = base + n * 576 + h * 48;
  const float SC = 0.14433756729740643f;   // 1/sqrt(48)
  float sc[4], mx = -1e30f;
  for (int m = 0; m < 4; ++m) {
    const bf16* kk = base + m * 576 + 192 + h * 48;
    float d = 0.f;
    for (int i = 0; i < 48; ++i) d += (float)q[i] * (float)kk[i];
    sc[m] = d * SC;
    mx = fmaxf(mx, sc[m]);
  }
  float se = 0.f;
  for (int m = 0; m < 4; ++m) { sc[m] = __expf(sc[m] - mx); se += sc[m]; }
  float inv = 1.f / se;
  bf16* o = ao + (b * 4 + n) * 192 + h * 48;
  for (int i = 0; i < 48; ++i) {
    float a = 0.f;
    for (int m = 0; m < 4; ++m) a += sc[m] * (float)base[m * 576 + 384 + h * 48 + i];
    o[i] = (bf16)(a * inv);
  }
}

// ---------------------------------------------------------------------------
// Fused SwiGLU gate/up: hsw = silu(xn2@Wg) * (xn2@Wu). Same M=64 scheme,
// dual B-fragments -> 8 WMMA per k-step per wave.
// ---------------------------------------------------------------------------
__global__ __launch_bounds__(128) void k_gemm_gu(
    const bf16* __restrict__ Xn2,          // B x 288
    const bf16* __restrict__ Wtg,          // 1152 x 288
    const bf16* __restrict__ Wtu,          // 1152 x 288
    bf16* __restrict__ hsw) {              // B x 1152
  __shared__ __align__(16) bf16 As[2][64 * KC];
  const int m0 = blockIdx.x * 64;
  const bf16* Az = Xn2 + (long long)m0 * 288;
  int wave = threadIdx.x >> 5, lane = threadIdx.x & 31;
  int n0 = blockIdx.y * 64 + wave * 16;
  v8f ag[4] = {}, au[4] = {};
  stage_a64(As[0], Az, 288);
  for (int c = 0; c < 3; ++c) {            // K = 288 = 3 * KC
    __syncthreads();
    if (c + 1 < 3) stage_a64(As[(c + 1) & 1], Az + (c + 1) * KC, 288);
    const bf16* ap = As[c & 1];
    const int kc = c * KC;
#pragma unroll
    for (int kk = 0; kk < KC / 32; ++kk) {
      v16bf bg = load_b_frag(Wtg, 288, n0, kc + kk * 32);
      v16bf bu = load_b_frag(Wtu, 288, n0, kc + kk * 32);
      v16bf a[4];
#pragma unroll
      for (int mt = 0; mt < 4; ++mt)
        a[mt] = load_a_frag(ap + mt * 16 * KC, KC, kk * 32);
#pragma unroll
      for (int mt = 0; mt < 4; ++mt) {
        ag[mt] = wmma_bf(a[mt], bg, ag[mt]);
        au[mt] = wmma_bf(a[mt], bu, au[mt]);
      }
    }
  }
  int col = n0 + (lane & 15);
#pragma unroll
  for (int mt = 0; mt < 4; ++mt) {
    int rbase = m0 + mt * 16 + ((lane >> 4) << 3);
#pragma unroll
    for (int i = 0; i < 8; ++i) {
      float g = ag[mt][i], u = au[mt][i];
      hsw[(long long)(rbase + i) * 1152 + col] = (bf16)(g * sigm(g) * u);
    }
  }
}

// ---------------------------------------------------------------------------
// Host-side orchestration (graph-capture safe: only kernel launches on stream)
// ---------------------------------------------------------------------------
extern "C" void kernel_launch(void* const* d_in, const int* in_sizes, int n_in,
                              void* d_out, int out_size, void* d_ws, size_t ws_size,
                              hipStream_t stream) {
  const float* x      = (const float*)d_in[0];
  const float* hst    = (const float*)d_in[1];
  const float* ln_g   = (const float*)d_in[2];
  const float* ln_b   = (const float*)d_in[3];
  const float* W_in   = (const float*)d_in[4];
  const float* b_in   = (const float*)d_in[5];
  const float* Wx     = (const float*)d_in[6];
  const float* bx     = (const float*)d_in[7];
  const float* Wh     = (const float*)d_in[8];
  const float* bh     = (const float*)d_in[9];
  const float* cbn_g  = (const float*)d_in[10];
  const float* cbn_b  = (const float*)d_in[11];
  const float* Wqkv   = (const float*)d_in[12];
  const float* Wo     = (const float*)d_in[13];
  const float* fln_g  = (const float*)d_in[14];
  const float* fln_b  = (const float*)d_in[15];
  const float* W1     = (const float*)d_in[16];
  const float* b1     = (const float*)d_in[17];
  const float* W2     = (const float*)d_in[18];
  const float* b2     = (const float*)d_in[19];
  const float* Wbm    = (const float*)d_in[20];
  const float* bbm    = (const float*)d_in[21];
  const float* fn_g   = (const float*)d_in[22];
  const float* fn_b   = (const float*)d_in[23];
  const float* Wg     = (const float*)d_in[24];
  const float* Wu     = (const float*)d_in[25];
  const float* Wd     = (const float*)d_in[26];
  const float* sscale = (const float*)d_in[27];

  const int B = in_sizes[0] / 288;        // 32768
  const long long BL = B;

  // --- workspace carve-out (256B aligned) ---
  char* ws = (char*)d_ws;
  size_t o = 0;
  auto carve = [&](size_t bytes) -> char* {
    char* p = ws + o;
    o = (o + bytes + 255) & ~(size_t)255;
    return p;
  };
  bf16* wt_in  = (bf16*)carve(768 * 288 * 2);
  bf16* wt_gru = (bf16*)carve(4 * 576 * 384 * 2);
  bf16* wt_hc  = (bf16*)carve(4 * 192 * 192 * 2);
  float* gbias = (float*)carve(4 * 768 * 4);
  bf16* wt_qkv = (bf16*)carve(576 * 192 * 2);
  bf16* wt_o   = (bf16*)carve(192 * 192 * 2);
  bf16* wt_1   = (bf16*)carve(4 * 384 * 192 * 2);
  bf16* wt_2   = (bf16*)carve(4 * 192 * 384 * 2);
  bf16* wt_bm  = (bf16*)carve(288 * 768 * 2);
  bf16* wt_g   = (bf16*)carve(1152 * 288 * 2);
  bf16* wt_u   = (bf16*)carve(1152 * 288 * 2);
  bf16* wt_d   = (bf16*)carve(288 * 1152 * 2);
  // activation regions with phase reuse
  char* R_qkv  = carve(BL * 2304 * 2);    // phase A: xn(+20MB bi); B: qkv; C: xm/xn2/hsw
  char* R_agru = carve(BL * 1536 * 2);    // Agru, later h1
  float* h_new = (float*)carve(BL * 768 * 4);
  bf16* sn_bf  = (bf16*)carve(BL * 768 * 2);
  char* R_ao   = carve(BL * 768 * 2);     // ao -> bn -> flat
  float* base  = (float*)carve(BL * 768 * 4);
  (void)ws_size; (void)n_in; (void)out_size;

  bf16* xn_bf  = (bf16*)(R_qkv);
  bf16* bi_bf  = (bf16*)(R_qkv + (20u << 20));
  bf16* qkv_bf = (bf16*)(R_qkv);
  float* xm    = (float*)(R_qkv);
  bf16* xn2_bf = (bf16*)(R_qkv + (40u << 20));
  bf16* hsw_bf = (bf16*)(R_qkv + (64u << 20));
  bf16* Agru   = (bf16*)R_agru;
  bf16* h1_bf  = (bf16*)R_agru;
  bf16* ao_bf  = (bf16*)R_ao;
  bf16* bn_bf  = (bf16*)R_ao;
  bf16* flat_bf= (bf16*)R_ao;

  float* out_x = (float*)d_out;                 // B x 288
  float* out_s = (float*)d_out + BL * 288;      // B x 4 x 192

  auto gwt = [](long long n) { return dim3((unsigned)((n + 255) / 256)); };

  // --- weight prep ---
  k_wt<<<gwt(768LL * 288), 256, 0, stream>>>(W_in, 768, wt_in, 288, 768);
  k_wt_gru<<<gwt(4LL * 576 * 384), 256, 0, stream>>>(Wx, Wh, wt_gru);
  for (int bnd = 0; bnd < 4; ++bnd) {
    k_wt<<<gwt(192LL * 192), 256, 0, stream>>>(Wh + bnd * 110592 + 384, 576,
                                               wt_hc + bnd * 36864, 192, 192);
    k_wt<<<gwt(384LL * 192), 256, 0, stream>>>(W1 + bnd * 73728, 384,
                                               wt_1 + bnd * 73728, 192, 384);
    k_wt<<<gwt(192LL * 384), 256, 0, stream>>>(W2 + bnd * 73728, 192,
                                               wt_2 + bnd * 73728, 384, 192);
  }
  k_gbias<<<gwt(4 * 768), 256, 0, stream>>>(bx, bh, gbias);
  k_wt<<<gwt(576LL * 192), 256, 0, stream>>>(Wqkv, 576, wt_qkv, 192, 576);
  k_wt<<<gwt(192LL * 192), 256, 0, stream>>>(Wo, 192, wt_o, 192, 192);
  k_wt<<<gwt(288LL * 768), 256, 0, stream>>>(Wbm, 288, wt_bm, 768, 288);
  k_wt<<<gwt(1152LL * 288), 256, 0, stream>>>(Wg, 1152, wt_g, 288, 1152);
  k_wt<<<gwt(1152LL * 288), 256, 0, stream>>>(Wu, 1152, wt_u, 288, 1152);
  k_wt<<<gwt(288LL * 1152), 256, 0, stream>>>(Wd, 288, wt_d, 1152, 288);

  const float* nofp = nullptr;
  bf16* nobf = nullptr;
  float* nof = nullptr;

  // 1) xn = LN(x)
  k_ln<<<dim3(B / 8), 256, 0, stream>>>(x, xn_bf, 288, 0, ln_g, ln_b, B);
  // 2) bi = xn @ W_in + b_in   (bf16 out)
  k_gemm<<<dim3(B / 64, 12, 1), 128, 0, stream>>>(
      xn_bf, 288, 0, wt_in, 0, b_in, 0, B, 768, 288, EPI_BF16, 0,
      nof, 0, 0, bi_bf, 768, 0, nofp, 0, 0, nofp);
  // 3) Agru = [bi | bf16(h_state)] per band
  k_pack<<<gwt(BL * 1536), 256, 0, stream>>>(bi_bf, hst, Agru, B);
  // 4) fused GRU -> h_new (f32)
  k_gru<<<dim3(B / 16, 1, 4), 256, 0, stream>>>(Agru, wt_gru, wt_hc, gbias, hst, h_new);
  // 5) sn = LN(h_new) with cbn
  k_ln<<<dim3(B * 4 / 8), 256, 0, stream>>>(h_new, sn_bf, 192, 0, cbn_g, cbn_b, B * 4);
  // 6) qkv = sn @ Wqkv
  k_gemm<<<dim3(B * 4 / 64, 9, 1), 128, 0, stream>>>(
      sn_bf, 192, 0, wt_qkv, 0, nofp, 0, B * 4, 576, 192, EPI_BF16, 0,
      nof, 0, 0, qkv_bf, 576, 0, nofp, 0, 0, nofp);
  // 7) tiny 4x4 cross-band attention
  k_attn<<<gwt(BL * 16), 256, 0, stream>>>(qkv_bf, ao_bf, B);
  // 8) base = h_new + ao @ Wo
  k_gemm<<<dim3(B * 4 / 64, 3, 1), 128, 0, stream>>>(
      ao_bf, 192, 0, wt_o, 0, nofp, 0, B * 4, 192, 192, EPI_F32ADD, 0,
      base, 192, 0, nobf, 0, 0, h_new, 192, 0, nofp);
  // 9) bn = LN(base) with per-band fln
  k_ln<<<dim3(B * 4 / 8), 256, 0, stream>>>(base, bn_bf, 192, 1, fln_g, fln_b, B * 4);
  // 10) h1 = silu(bn @ W1 + b1)  per band
  k_gemm<<<dim3(B / 64, 6, 4), 128, 0, stream>>>(
      bn_bf, 768, 192, wt_1, 384 * 192, b1, 384, B, 384, 192, EPI_BF16, 1,
      nof, 0, 0, h1_bf, 1536, 384, nofp, 0, 0, nofp);
  // 11) f = h1 @ W2 + b2; flat = bf16(base+f); new_states = base + f*scale
  k_gemm<<<dim3(B / 64, 3, 4), 128, 0, stream>>>(
      h1_bf, 1536, 384, wt_2, 192 * 384, b2, 192, B, 192, 384, EPI_DUAL, 0,
      out_s, 768, 192, flat_bf, 768, 192, base, 768, 192, sscale);
  // 12) xm = x + flat @ Wbm + bbm
  k_gemm<<<dim3(B / 64, 5, 1), 128, 0, stream>>>(
      flat_bf, 768, 0, wt_bm, 0, bbm, 0, B, 288, 768, EPI_F32ADD, 0,
      xm, 288, 0, nobf, 0, 0, x, 288, 0, nofp);
  // 13) xn2 = LN(xm)
  k_ln<<<dim3(B / 8), 256, 0, stream>>>(xm, xn2_bf, 288, 0, fn_g, fn_b, B);
  // 14) hsw = silu(xn2@Wg) * (xn2@Wu)
  k_gemm_gu<<<dim3(B / 64, 18, 1), 128, 0, stream>>>(xn2_bf, wt_g, wt_u, hsw_bf);
  // 15) out = xm + hsw @ Wd
  k_gemm<<<dim3(B / 64, 5, 1), 128, 0, stream>>>(
      hsw_bf, 1152, 0, wt_d, 0, nofp, 0, B, 288, 1152, EPI_F32ADD, 0,
      out_x, 288, 0, nobf, 0, 0, xm, 288, 0, nofp);
}
